// SpanClassifier_32255204393722
// MI455X (gfx1250) — compile-verified, matrix operand order checked by reading, not run on hardware
//
#include <hip/hip_runtime.h>
#include <hip/hip_bf16.h>

// ---------------------------------------------------------------------------
// CDNA5 (gfx1250) span-classifier. All large matmuls go through
// v_wmma_f32_16x16x32_bf16 with double-buffered LDS fed by
// GLOBAL_LOAD_ASYNC_TO_LDS_B128 (ASYNCcnt-tracked async copies).
// Every GEMM is NT (C = A * B^T, both row-major, K-contiguous), so async
// staging is pure 64B row-segment copies; constant matrices (hid, wq, wk,
// sp_w) are transposed/converted to bf16 once per call.
// ---------------------------------------------------------------------------

typedef __attribute__((ext_vector_type(16))) __bf16 v16bf;
typedef __attribute__((ext_vector_type(8)))  float  v8f;

__device__ __forceinline__ unsigned short f2bf(float x) {
  unsigned int u = __float_as_uint(x);
  u += 0x7FFFu + ((u >> 16) & 1u);          // round-to-nearest-even
  return (unsigned short)(u >> 16);
}
__device__ __forceinline__ float bf2f(unsigned short h) {
  return __uint_as_float(((unsigned int)h) << 16);
}

// ------------------------------- GEMM --------------------------------------
// C[M,N] = act( scale * (A[M,K] * Bt[N,K]^T + bias) )
// A may be split at K0 between A0/A1 (for the concat([hidp,hidc]) @ sp_w GEMM).
struct GemmP {
  const unsigned short* A0;
  const unsigned short* A1;    // nullptr => no split
  const unsigned short* Bt;    // [N,K] row-major  (== B^T)
  const float* bias;           // [N] or nullptr
  float* Cf;                   // f32 out or nullptr
  unsigned short* Cbf;         // bf16 out or nullptr
  int M, N, K, K0;
  int ldA, ldB, ldC;
  long long sA, sB, sC;        // per-batch element strides (0 = shared)
  float scale;
  int relu;
};

#define BM 128
#define BN 128
#define BK 32
#define LSTR 40                // LDS row stride in bf16 elems (80B, 16B-aligned)

// 4 async b128 copies = one 64B row segment (offset applies to LDS AND global)
#define ASYNC_ROW64(lds, voff, sbase)                                       \
  asm volatile("global_load_async_to_lds_b128 %0, %1, %2 offset:0\n\t"      \
               "global_load_async_to_lds_b128 %0, %1, %2 offset:16\n\t"     \
               "global_load_async_to_lds_b128 %0, %1, %2 offset:32\n\t"     \
               "global_load_async_to_lds_b128 %0, %1, %2 offset:48"         \
               :: "v"(lds), "v"(voff), "s"(sbase) : "memory")

__global__ __launch_bounds__(128) void gemm_bf16_nt(GemmP p) {
  // double-buffered tiles, rows 80B so all b128 LDS addresses are 16B aligned
  __shared__ __align__(16) unsigned short lA[2 * BM * LSTR];
  __shared__ __align__(16) unsigned short lB[2 * BN * LSTR];

  const int bz  = blockIdx.z;
  const int bm0 = blockIdx.y * BM;
  const int bn0 = blockIdx.x * BN;
  const int tid  = threadIdx.x;          // 128 threads = 4 waves
  const int lane = tid & 31;
  const int wid  = tid >> 5;
  const int wm = wid >> 1;               // 0..1 : 64-row slab
  const int wn = wid & 1;                // 0..1 : 64-col slab
  const int ml = lane & 15;
  const int hh = lane >> 4;

  const unsigned short* A0 = p.A0 + (long long)bz * p.sA;
  const unsigned short* A1 = p.A1 ? (p.A1 + (long long)bz * p.sA) : nullptr;
  const unsigned short* Bt = p.Bt + (long long)bz * p.sB;

  // LDS byte addresses for this thread's staging slot (one 64B row per tile)
  const unsigned ldsAbase = (unsigned)(size_t)(&lA[0]) + (unsigned)tid * (LSTR * 2);
  const unsigned ldsBbase = (unsigned)(size_t)(&lB[0]) + (unsigned)tid * (LSTR * 2);
  const unsigned aRowG = (unsigned)((bm0 + tid) * p.ldA) * 2u;  // global row byte off
  const unsigned bRowG = (unsigned)((bn0 + tid) * p.ldB) * 2u;

  const int nkt = p.K / BK;

  auto issue = [&](int kt) {
    const int gk = kt * BK;
    const unsigned short* Asrc = A0;
    int acol = gk;
    if (A1 && gk >= p.K0) { Asrc = A1; acol = gk - p.K0; }
    const unsigned buf = (unsigned)(kt & 1) * (BM * LSTR * 2);
    ASYNC_ROW64(ldsAbase + buf, aRowG + (unsigned)acol * 2u,
                (unsigned long long)(size_t)Asrc);
    ASYNC_ROW64(ldsBbase + buf, bRowG + (unsigned)gk * 2u,
                (unsigned long long)(size_t)Bt);
  };

  v8f acc[4][4] = {};

  issue(0);
  if (nkt > 1) issue(1);

  for (int kt = 0; kt < nkt; ++kt) {
    // tile kt's 8 async ops are the oldest; leave the next tile's 8 in flight
    if (kt + 1 < nkt) asm volatile("s_wait_asynccnt 0x8" ::: "memory");
    else              asm volatile("s_wait_asynccnt 0x0" ::: "memory");
    __syncthreads();

    const unsigned short* bufA = &lA[(kt & 1) * (BM * LSTR)];
    const unsigned short* bufB = &lB[(kt & 1) * (BN * LSTR)];

    // fragments per ISA 16-bit 16x32 layout:
    // lanes 0-15: M=0..15, VGPR j<4 -> K=2j,2j+1 ; j>=4 -> K=8+2j,9+2j
    // lanes16-31: same +8 in K
    union FU { v16bf v; unsigned int u[8]; };
    v16bf afr[4], bfr[4];
#pragma unroll
    for (int tm = 0; tm < 4; ++tm) {
      const unsigned short* base = &bufA[(wm * 64 + tm * 16 + ml) * LSTR];
      FU f;
#pragma unroll
      for (int j = 0; j < 8; ++j) {
        const int koff = (j < 4) ? (2 * j + 8 * hh) : (8 + 2 * j + 8 * hh);
        f.u[j] = *reinterpret_cast<const unsigned int*>(base + koff);
      }
      afr[tm] = f.v;
    }
#pragma unroll
    for (int tn = 0; tn < 4; ++tn) {
      const unsigned short* base = &bufB[(wn * 64 + tn * 16 + ml) * LSTR];
      FU f;
#pragma unroll
      for (int j = 0; j < 8; ++j) {
        const int koff = (j < 4) ? (2 * j + 8 * hh) : (8 + 2 * j + 8 * hh);
        f.u[j] = *reinterpret_cast<const unsigned int*>(base + koff);
      }
      bfr[tn] = f.v;
    }
#pragma unroll
    for (int tm = 0; tm < 4; ++tm)
#pragma unroll
      for (int tn = 0; tn < 4; ++tn)
        acc[tm][tn] = __builtin_amdgcn_wmma_f32_16x16x32_bf16(
            false, afr[tm], false, bfr[tn], (short)0, acc[tm][tn], false, false);

    __syncthreads();                 // all reads of buf[kt&1] finished
    if (kt + 2 < nkt) issue(kt + 2); // safe to overwrite it now
  }

  float* Cf = p.Cf ? p.Cf + (long long)bz * p.sC : nullptr;
  unsigned short* Cb = p.Cbf ? p.Cbf + (long long)bz * p.sC : nullptr;
#pragma unroll
  for (int tm = 0; tm < 4; ++tm) {
#pragma unroll
    for (int tn = 0; tn < 4; ++tn) {
      const int col = bn0 + wn * 64 + tn * 16 + ml;
      const float bval = p.bias ? p.bias[col] : 0.f;
#pragma unroll
      for (int e = 0; e < 8; ++e) {
        // C layout: VGPR e, lanes0-15 -> M=e ; lanes16-31 -> M=8+e
        const int row = bm0 + wm * 64 + tm * 16 + 8 * hh + e;
        float v = (acc[tm][tn][e] + bval) * p.scale;
        if (p.relu) v = fmaxf(v, 0.f);
        const long long idx = (long long)row * p.ldC + col;
        if (Cf) Cf[idx] = v;
        if (Cb) Cb[idx] = f2bf(v);
      }
    }
  }
}

// --------------------------- helper kernels --------------------------------

__global__ void conv_f32_bf16(const float* __restrict__ src,
                              unsigned short* __restrict__ d0,
                              unsigned short* __restrict__ d1, long long n) {
  long long i = (long long)blockIdx.x * blockDim.x + threadIdx.x;
  const long long stride = (long long)gridDim.x * blockDim.x;
  for (; i < n; i += stride) {
    const unsigned short v = f2bf(src[i]);
    d0[i] = v;
    if (d1) d1[i] = v;
  }
}

// dst[C,R] (bf16) = transpose(src[R,C] (f32)); dims multiples of 32
__global__ __launch_bounds__(256) void transp_f32_bf16(
    const float* __restrict__ src, unsigned short* __restrict__ dst,
    int R, int C, long long sSrc, long long sDst) {
  __shared__ float tile[32][33];
  const int b = blockIdx.z;
  src += (long long)b * sSrc;
  dst += (long long)b * sDst;
  const int c0 = blockIdx.x * 32, r0 = blockIdx.y * 32;
  const int tx = threadIdx.x, ty = threadIdx.y;     // block (32,8)
#pragma unroll
  for (int j = 0; j < 32; j += 8)
    tile[ty + j][tx] = src[(long long)(r0 + ty + j) * C + (c0 + tx)];
  __syncthreads();
#pragma unroll
  for (int j = 0; j < 32; j += 8)
    dst[(long long)(c0 + ty + j) * R + (r0 + tx)] = f2bf(tile[tx][ty + j]);
}

// attn_w0 = amask / clip(rowsum, 1e-6) -> bf16 into both chains' attn buffers
__global__ __launch_bounds__(256) void attn0_kernel(
    const unsigned char* __restrict__ am, const unsigned char* __restrict__ sm,
    unsigned short* __restrict__ a_st, unsigned short* __restrict__ a_ed, int L) {
  __shared__ float red[256];
  const int b = blockIdx.y, q = blockIdx.x, tid = threadIdx.x;
  const unsigned char* amr = am + (long long)b * L;
  const int srcm = sm[(long long)b * L + q] ? 1 : 0;
  float c = 0.f;
  for (int k = tid; k < L; k += 256) c += amr[k] ? 1.f : 0.f;
  red[tid] = c; __syncthreads();
  for (int s = 128; s > 0; s >>= 1) {
    if (tid < s) red[tid] += red[tid + s];
    __syncthreads();
  }
  const float sum = srcm ? red[0] : 0.f;
  const float inv = 1.f / fmaxf(sum, 1e-6f);
  const long long base = ((long long)b * L + q) * L;
  for (int k = tid; k < L; k += 256) {
    const unsigned short v = f2bf((srcm && amr[k]) ? inv : 0.f);
    a_st[base + k] = v;
    a_ed[base + k] = v;
  }
}

// qrel[b,q,r] = sum_d q_bf16[b,q,d] * rel[r,d]   (r in 0..32)
__global__ __launch_bounds__(128) void qrel_kernel(
    const unsigned short* __restrict__ qbf, const float* __restrict__ rel,
    float* __restrict__ qrel, int D, int L) {
  __shared__ float qrow[1024];
  const int b = blockIdx.y, q = blockIdx.x, tid = threadIdx.x;
  const long long base = ((long long)b * L + q) * D;
  for (int d = tid; d < D; d += 128) qrow[d] = bf2f(qbf[base + d]);
  __syncthreads();
  if (tid < 33) {
    const float* r = rel + (long long)tid * D;
    float s = 0.f;
    for (int d = 0; d < D; ++d) s += qrow[d] * r[d];
    qrel[((long long)b * L + q) * 33 + tid] = s;
  }
}

// masked softmax over k of sc + gather(qrel); writes out[b,q,i,:] f32 and
// bf16 attention for the next iteration's attn @ hid GEMM.
__global__ __launch_bounds__(256) void softmax_kernel(
    const float* __restrict__ sc, const float* __restrict__ qrel,
    const int* __restrict__ spw, const unsigned char* __restrict__ am,
    const unsigned char* __restrict__ sm, float* __restrict__ out,
    unsigned short* __restrict__ attn_bf, int i, int L, int S, int Mrel) {
  __shared__ float red[256];
  const int b = blockIdx.y, q = blockIdx.x, tid = threadIdx.x;
  const long long rq = (long long)b * L + q;
  const float* scrow = sc + rq * L;
  const float* qr = qrel + rq * 33;
  const bool mq = (i < spw[rq]) && sm[rq];
  const unsigned char* amr = am + (long long)b * L;

  float vals[4];
  float mx = -1e30f;
#pragma unroll
  for (int t = 0; t < 4; ++t) {
    const int k = tid + t * 256;
    int r = k - q; r = (r < -Mrel) ? -Mrel : (r > Mrel ? Mrel : r);
    const bool mk = mq && amr[k];
    const float v = mk ? (scrow[k] + qr[r + Mrel]) : -1e18f;
    vals[t] = v;
    mx = fmaxf(mx, v);
  }
  red[tid] = mx; __syncthreads();
  for (int s = 128; s > 0; s >>= 1) {
    if (tid < s) red[tid] = fmaxf(red[tid], red[tid + s]);
    __syncthreads();
  }
  mx = red[0]; __syncthreads();
  float sum = 0.f;
#pragma unroll
  for (int t = 0; t < 4; ++t) {
    vals[t] = __expf(vals[t] - mx);
    sum += vals[t];
  }
  red[tid] = sum; __syncthreads();
  for (int s = 128; s > 0; s >>= 1) {
    if (tid < s) red[tid] += red[tid + s];
    __syncthreads();
  }
  const float inv = 1.f / red[0];
  float* orow = out + (rq * S + i) * (long long)L;
  unsigned short* arow = attn_bf + rq * L;
#pragma unroll
  for (int t = 0; t < 4; ++t) {
    const int k = tid + t * 256;
    const float w = vals[t] * inv;
    orow[k] = w;
    arow[k] = f2bf(w);
  }
}

__global__ void masks_kernel(const int* __restrict__ spw,
                             float* __restrict__ out, int n, int S) {
  const int idx = blockIdx.x * blockDim.x + threadIdx.x;
  if (idx >= n * S) return;
  const int s = idx % S, row = idx / S;
  out[idx] = (s < spw[row]) ? 1.f : 0.f;
}

// ------------------------------ host side ----------------------------------

extern "C" void kernel_launch(void* const* d_in, const int* in_sizes, int n_in,
                              void* d_out, int out_size, void* d_ws, size_t ws_size,
                              hipStream_t stream) {
  (void)in_sizes; (void)n_in; (void)out_size;
  const float* hid   = (const float*)d_in[0];
  const float* srch  = (const float*)d_in[1];
  const int*   spw   = (const int*)d_in[2];
  const unsigned char* am = (const unsigned char*)d_in[4];
  const unsigned char* sm = (const unsigned char*)d_in[5];
  const float* wq_st = (const float*)d_in[6];
  const float* bq_st = (const float*)d_in[7];
  const float* wk_st = (const float*)d_in[8];
  const float* bk_st = (const float*)d_in[9];
  const float* rel_st = (const float*)d_in[10];
  const float* wq_ed = (const float*)d_in[11];
  const float* bq_ed = (const float*)d_in[12];
  const float* wk_ed = (const float*)d_in[13];
  const float* bk_ed = (const float*)d_in[14];
  const float* rel_ed = (const float*)d_in[15];
  const float* sp_w  = (const float*)d_in[16];
  const float* sp_b  = (const float*)d_in[17];

  constexpr int B = 4, L = 1024, D = 1024, S = 8, MR = 16;
  constexpr long long BLD = (long long)B * L * D;
  constexpr long long BLL = (long long)B * L * L;
  constexpr long long DD  = (long long)D * D;
  const float qscale = 0.03125f;   // 1/sqrt(1024)

  char* wsp = (char*)d_ws;
  auto carve = [&](long long bytes) -> void* {
    void* p = (void*)wsp;
    wsp += (bytes + 255) & ~255ll;
    return p;
  };
  unsigned short* hid_bf  = (unsigned short*)carve(BLD * 2);
  unsigned short* hidT_bf = (unsigned short*)carve(BLD * 2);     // [B, D, L]
  unsigned short* h1[2] = {(unsigned short*)carve(BLD * 2), (unsigned short*)carve(BLD * 2)};
  unsigned short* h2[2] = {(unsigned short*)carve(BLD * 2), (unsigned short*)carve(BLD * 2)};
  unsigned short* wqstT = (unsigned short*)carve(DD * 2);
  unsigned short* wkstT = (unsigned short*)carve(DD * 2);
  unsigned short* wqedT = (unsigned short*)carve(DD * 2);
  unsigned short* wkedT = (unsigned short*)carve(DD * 2);
  unsigned short* spwT  = (unsigned short*)carve(2 * DD * 2);    // [D, 2D]
  unsigned short* kst   = (unsigned short*)carve(BLD * 2);
  unsigned short* ked   = (unsigned short*)carve(BLD * 2);
  unsigned short* a_st  = (unsigned short*)carve(BLL * 2);
  unsigned short* a_ed  = (unsigned short*)carve(BLL * 2);
  unsigned short* hidc  = (unsigned short*)carve(BLD * 2);
  unsigned short* qbf   = (unsigned short*)carve(BLD * 2);
  float* qrelf = (float*)carve((long long)B * L * 33 * 4);
  float* scf   = (float*)carve(BLL * 4);
  if ((size_t)(wsp - (char*)d_ws) > ws_size) return;  // workspace too small

  float* out = (float*)d_out;
  float* sts_out = out;
  float* eds_out = out + (long long)B * L * S * L;
  float* msk_out = out + 2ll * B * L * S * L;

  // ---- one-time (per call) prep ----
  conv_f32_bf16<<<1024, 256, 0, stream>>>(hid, hid_bf, nullptr, BLD);
  conv_f32_bf16<<<1024, 256, 0, stream>>>(srch, h1[0], h2[0], BLD);
  transp_f32_bf16<<<dim3(D / 32, L / 32, B), dim3(32, 8), 0, stream>>>(
      hid, hidT_bf, L, D, (long long)L * D, (long long)D * L);
  transp_f32_bf16<<<dim3(D / 32, D / 32, 1), dim3(32, 8), 0, stream>>>(wq_st, wqstT, D, D, 0, 0);
  transp_f32_bf16<<<dim3(D / 32, D / 32, 1), dim3(32, 8), 0, stream>>>(wk_st, wkstT, D, D, 0, 0);
  transp_f32_bf16<<<dim3(D / 32, D / 32, 1), dim3(32, 8), 0, stream>>>(wq_ed, wqedT, D, D, 0, 0);
  transp_f32_bf16<<<dim3(D / 32, D / 32, 1), dim3(32, 8), 0, stream>>>(wk_ed, wkedT, D, D, 0, 0);
  transp_f32_bf16<<<dim3(D / 32, 2 * D / 32, 1), dim3(32, 8), 0, stream>>>(sp_w, spwT, 2 * D, D, 0, 0);
  attn0_kernel<<<dim3(L, B), 256, 0, stream>>>(am, sm, a_st, a_ed, L);
  masks_kernel<<<(B * L * S + 255) / 256, 256, 0, stream>>>(spw, msk_out, B * L, S);

  auto gemm = [&](const unsigned short* A0, const unsigned short* A1,
                  const unsigned short* Bt, const float* bias, float* Cf,
                  unsigned short* Cbf, int M, int N, int K, int K0,
                  int ldA, int ldB, long long sA, long long sB, long long sC,
                  float scale, int relu) {
    GemmP p{A0, A1, Bt, bias, Cf, Cbf, M, N, K, K0, ldA, ldB, N, sA, sB, sC, scale, relu};
    gemm_bf16_nt<<<dim3(N / BN, M / BM, B), 128, 0, stream>>>(p);
  };

  // k projections are iteration-invariant: k = hid @ wk + bk
  gemm(hid_bf, nullptr, wkstT, bk_st, nullptr, kst, L, D, D, D, D, D,
       (long long)L * D, 0, (long long)L * D, 1.f, 0);
  gemm(hid_bf, nullptr, wkedT, bk_ed, nullptr, ked, L, D, D, D, D, D,
       (long long)L * D, 0, (long long)L * D, 1.f, 0);

  int p1 = 0, p2 = 0;  // ping-pong indices for the two chains
  for (int i = 0; i < S; ++i) {
    for (int c = 0; c < 2; ++c) {
      unsigned short** hp   = (c == 0) ? h1 : h2;
      int&             ping = (c == 0) ? p1 : p2;
      unsigned short*  attn = (c == 0) ? a_st : a_ed;
      unsigned short*  kmat = (c == 0) ? kst : ked;
      const unsigned short* wqT = (c == 0) ? wqstT : wqedT;
      const float* bq  = (c == 0) ? bq_st : bq_ed;
      const float* rel = (c == 0) ? rel_st : rel_ed;
      float* obase     = (c == 0) ? sts_out : eds_out;

      // hidc = attn_prev @ hid            [L,L] x [L,D]
      gemm(attn, nullptr, hidT_bf, nullptr, nullptr, hidc, L, D, L, L, L, L,
           BLL / B, (long long)D * L, (long long)L * D, 1.f, 0);
      // hid_new = relu(concat(hidp,hidc) @ sp_w + sp_b)   K = 2D, split at D
      gemm(hp[ping], hidc, spwT, sp_b, nullptr, hp[ping ^ 1], L, D, 2 * D, D,
           D, 2 * D, (long long)L * D, 0, (long long)L * D, 1.f, 1);
      ping ^= 1;
      // q = (hid_new @ wq + bq) * 1/sqrt(D)
      gemm(hp[ping], nullptr, wqT, bq, nullptr, qbf, L, D, D, D, D, D,
           (long long)L * D, 0, (long long)L * D, qscale, 0);
      // qrel = q @ rel^T  [L, 33]
      qrel_kernel<<<dim3(L, B), 128, 0, stream>>>(qbf, rel, qrelf, D, L);
      // sc = q @ k^T  [L, L]
      gemm(qbf, nullptr, kmat, nullptr, scf, nullptr, L, L, D, D, D, D,
           (long long)L * D, (long long)L * D, BLL / B, 1.f, 0);
      // masked softmax(sc + gather(qrel)) -> output slice i + next attn (bf16)
      softmax_kernel<<<dim3(L, B), 256, 0, stream>>>(
          scf, qrelf, spw, am, sm, obase, attn, i, L, S, MR);
    }
  }
}
